// CCMoE_19885698580851
// MI455X (gfx1250) — compile-verified, matrix operand order checked by reading, not run on hardware
//
#include <hip/hip_runtime.h>
#include <cstdint>

typedef __attribute__((ext_vector_type(16))) _Float16 v16h;
typedef __attribute__((ext_vector_type(8)))  float    v8f;
typedef __attribute__((ext_vector_type(4)))  int      v4i;

#define HID_    256
#define M_      16
#define ACT_    6
#define R_      24.0f
#define NEG_    -1e9f

#define AS1 __attribute__((address_space(1)))
#define AS3 __attribute__((address_space(3)))

#if defined(__has_builtin)
#  if __has_builtin(__builtin_amdgcn_global_load_async_to_lds_b128)
#    define HAVE_ASYNC_LDS 1
#  endif
#endif
#ifndef HAVE_ASYNC_LDS
#  define HAVE_ASYNC_LDS 0
#endif

__device__ __forceinline__ void wait_async0() {
#if defined(__has_builtin) && __has_builtin(__builtin_amdgcn_s_wait_asynccnt)
  __builtin_amdgcn_s_wait_asynccnt(0);
#else
  asm volatile("s_wait_asynccnt 0" ::: "memory");
#endif
}

// Branch-free tanh: (e^2x - 1) * rcp(e^2x + 1); clamp keeps exp finite.
__device__ __forceinline__ float fast_tanh(float x) {
  const float cx = fminf(fmaxf(x, -15.0f), 15.0f);
  const float e  = __expf(2.0f * cx);
  return (e - 1.0f) * __builtin_amdgcn_rcpf(e + 1.0f);
}

union Frag { v16h h; uint32_t u[8]; };

// 16x32 f16 fragment (ISA 7.12.2 A-layout): lane l -> row l&15, K-half l>>4;
// dwords u[0..3] and u[4..7] are two contiguous 16B runs -> b128-mergeable.
__device__ __forceinline__ void load_frag(Frag& f, const uint32_t* rowp, int kt, int half4) {
#pragma unroll
  for (int j = 0; j < 4; ++j) {
    f.u[j]     = rowp[kt * 16 + half4 + j];
    f.u[4 + j] = rowp[kt * 16 + 8 + half4 + j];
  }
}

// C = act(A[B,K] @ Wt[N,K]^T + bias). One wave per 16-row stripe, 8 waves/block.
// A-fragments register-cached for the whole N loop; weight panel staged into
// LDS (async-to-LDS when available) in <=64KB chunks shared by all 8 waves.
template<int KT, int NT, bool DOTANH, bool OUTF32>
__global__ __launch_bounds__(256) void gemm_wmma_kern(
    const _Float16* __restrict__ A, const _Float16* __restrict__ Wt,
    const float* __restrict__ bias, void* __restrict__ outp, int N)
{
  constexpr int K   = KT * 32;
  constexpr int CNT = (NT * KT <= 64) ? NT : (64 / KT);   // n-tiles per LDS chunk
  static_assert(NT % CNT == 0, "NT must be a multiple of chunk size");
  constexpr int CHUNK_BYTES = CNT * 16 * K * 2;           // <= 64KB
  __shared__ __align__(16) _Float16 smem[CNT * 16 * K];

  const int lane    = threadIdx.x & 31;
  const int wave    = threadIdx.x >> 5;
  const int rowbase = (blockIdx.x * 8 + wave) * 16;
  const int mrow    = lane & 15;
  const int half4   = (lane >> 4) << 2;

  Frag af[KT];
  const uint32_t* arow = (const uint32_t*)(A + (size_t)(rowbase + mrow) * K);
  __builtin_prefetch(arow, 0, 1);
#pragma unroll
  for (int kt = 0; kt < KT; ++kt) load_frag(af[kt], arow, kt, half4);

  for (int nc = 0; nc < NT; nc += CNT) {
    if (nc) __syncthreads();   // protect previous chunk's readers
    // ---- cooperative stage of weight panel rows [nc*16, (nc+CNT)*16) ----
    char* src = const_cast<char*>((const char*)(Wt + (size_t)nc * 16 * K));
    char* dst = (char*)smem;
    for (int off = threadIdx.x * 16; off < CHUNK_BYTES; off += 256 * 16) {
#if HAVE_ASYNC_LDS
      __builtin_amdgcn_global_load_async_to_lds_b128(
          (AS1 v4i*)(src + off), (AS3 v4i*)(dst + off), 0, 0);
#else
      *(uint4*)(dst + off) = *(const uint4*)(src + off);
#endif
    }
#if HAVE_ASYNC_LDS
    wait_async0();
#endif
    __syncthreads();

#pragma unroll
    for (int nt2 = 0; nt2 < CNT; ++nt2) {
      const int nt = nc + nt2;
      const float bn = bias[nt * 16 + mrow];
      v8f acc = { bn, bn, bn, bn, bn, bn, bn, bn };
      const uint32_t* wrow = (const uint32_t*)(smem + (size_t)(nt2 * 16 + mrow) * K);
#pragma unroll
      for (int kt = 0; kt < KT; ++kt) {
        Frag bf;
        load_frag(bf, wrow, kt, half4);   // 2x ds_load_b128
        acc = __builtin_amdgcn_wmma_f32_16x16x32_f16(false, af[kt].h, false, bf.h,
                                                     (short)0, acc, false, false);
      }
      // C/D layout: VGPR r, lanes 0-15 -> M=r, lanes 16-31 -> M=r+8; N = lane&15.
      const int col = nt * 16 + mrow;
      const int rb  = rowbase + ((lane >> 4) << 3);
#pragma unroll
      for (int r = 0; r < 8; ++r) {
        float v = acc[r];
        if (DOTANH) v = fast_tanh(v);
        const size_t idx = (size_t)(rb + r) * N + col;
        if (OUTF32) ((float*)outp)[idx]    = v;
        else        ((_Float16*)outp)[idx] = (_Float16)v;
      }
    }
  }
}

// obs [B,17] f32 -> [B,32] f16, zero padded
__global__ __launch_bounds__(256) void prep_obs(const float* __restrict__ obs,
                                                _Float16* __restrict__ out, int total)
{
  for (int i = blockIdx.x * blockDim.x + threadIdx.x; i < total;
       i += gridDim.x * blockDim.x) {
    const int b = i >> 5, k = i & 31;
    out[i] = (_Float16)((k < 17) ? obs[b * 17 + k] : 0.0f);
  }
}

// Build all transposed f16 weights + center stats. 65536 threads, each index once.
__global__ __launch_bounds__(256) void prep_weights(
    const float* __restrict__ W1,  const float* __restrict__ W2,
    const float* __restrict__ Wg1, const float* __restrict__ Wg2,
    const float* __restrict__ muW, const float* __restrict__ Wv1,
    const float* __restrict__ Wv2, const float* __restrict__ centers,
    const float* __restrict__ mean, const float* __restrict__ var,
    _Float16* Wt1, _Float16* Wt2, _Float16* Wtg1, _Float16* Wtg2,
    _Float16* WtMU, _Float16* WtV1, _Float16* WtV2, _Float16* WtC,
    float* cc, float* offm)
{
  const int t = blockIdx.x * blockDim.x + threadIdx.x;
  if (t < 65536) {                       // 256x256 transposes
    const int n = t >> 8, k = t & 255;
    Wt2[t]  = (_Float16)W2 [k * 256 + n];
    WtV1[t] = (_Float16)Wv1[k * 256 + n];
    WtV2[t] = (_Float16)Wv2[k * 256 + n];
  }
  if (t < 8192) {                        // W1: [17,256] -> [256,32] padded
    const int n = t >> 5, k = t & 31;
    Wt1[t] = (_Float16)((k < 17) ? W1[k * 256 + n] : 0.0f);
  }
  if (t < 32768) {                       // Wg1: [256,128] -> [128,256]
    const int n = t >> 8, k = t & 255;
    Wtg1[t] = (_Float16)Wg1[k * 128 + n];
  }
  if (t < 2048) {                        // Wg2: [128,16] -> [16,128]
    const int n = t >> 7, k = t & 127;
    Wtg2[t] = (_Float16)Wg2[k * 16 + n];
  }
  if (t < 24576) {                       // muW: [16,256,6] -> [96,256]
    const int n = t / 256, k = t % 256;
    const int m = n / 6, a = n - m * 6;
    WtMU[t] = (_Float16)muW[(m * 256 + k) * 6 + a];
  }
  if (t < 4096) {                        // whitened centers [16,256]
    const int m = t >> 8, k = t & 255;
    WtC[t] = (_Float16)(centers[m * 256 + k] * __builtin_amdgcn_rsqf(var[k] + 1e-6f));
  }
  if (t < 16) {                          // ||c||^2, -mean . (c/sig)
    float s1 = 0.f, s2 = 0.f;
    for (int k = 0; k < 256; ++k) {
      const float c  = centers[t * 256 + k];
      const float is = __builtin_amdgcn_rsqf(var[k] + 1e-6f);
      s1 += c * c;
      s2 += mean[k] * c * is;
    }
    cc[t]   = s1;
    offm[t] = -s2;
  }
}

// One wave per row: distances, masked softmax over 16 charts, combine, value dot.
__global__ __launch_bounds__(256) void epilogue_kern(
    const _Float16* __restrict__ FEAT, const float* __restrict__ LOGITS,
    const float* __restrict__ CD,      const _Float16* __restrict__ MUS,
    const _Float16* __restrict__ VH2,  const float* __restrict__ cc,
    const float* __restrict__ mean,    const float* __restrict__ var,
    const float* __restrict__ log_std, const float* __restrict__ Wv3,
    const float* __restrict__ bv3,
    float* __restrict__ out_mu, float* __restrict__ out_s, float* __restrict__ out_v)
{
  const int lane = threadIdx.x & 31;
  const int wave = threadIdx.x >> 5;
  const int row  = blockIdx.x * 8 + wave;

  // ||zw||^2 and v = VH2 . Wv3 (32-lane reductions)
  float zwsq = 0.f, vacc = 0.f;
  for (int d = lane; d < HID_; d += 32) {
    const float f = (float)FEAT[(size_t)row * HID_ + d];
    const float z = (f - mean[d]) * __builtin_amdgcn_rsqf(var[d] + 1e-6f);
    zwsq += z * z;
    vacc += (float)VH2[(size_t)row * HID_ + d] * Wv3[d];
  }
  for (int s = 16; s >= 1; s >>= 1) {
    zwsq += __shfl_xor(zwsq, s, 32);
    vacc += __shfl_xor(vacc, s, 32);
  }

  // lanes 0-15 own one chart each (16-31 mirror; xor-reductions stay in-half)
  const int m = lane & 15;
  const float d2    = fmaxf(zwsq + cc[m] - 2.f * CD[(size_t)row * M_ + m], 0.f);
  const float dist  = sqrtf(d2);
  const float logit = LOGITS[(size_t)row * M_ + m] - d2 * (1.0f / (R_ * R_));

  float dmin = dist;
  for (int s = 8; s >= 1; s >>= 1) dmin = fminf(dmin, __shfl_xor(dmin, s, 32));
  const bool active = (dist <= R_) || (dist == dmin);

  float x = active ? logit : NEG_;
  float xmax = x;
  for (int s = 8; s >= 1; s >>= 1) xmax = fmaxf(xmax, __shfl_xor(xmax, s, 32));
  const float e = __expf(x - xmax);
  float esum = e;
  for (int s = 8; s >= 1; s >>= 1) esum += __shfl_xor(esum, s, 32);
  const float g = e * __builtin_amdgcn_rcpf(esum);

  float pm[ACT_], ps[ACT_];
#pragma unroll
  for (int a = 0; a < ACT_; ++a) {
    pm[a] = g * (float)MUS[(size_t)row * 96 + m * ACT_ + a];
    ps[a] = g * log_std[m * ACT_ + a];
  }
  for (int s = 8; s >= 1; s >>= 1) {
#pragma unroll
    for (int a = 0; a < ACT_; ++a) {
      pm[a] += __shfl_xor(pm[a], s, 32);
      ps[a] += __shfl_xor(ps[a], s, 32);
    }
  }
  if (lane == 0) {
#pragma unroll
    for (int a = 0; a < ACT_; ++a) {
      out_mu[(size_t)row * ACT_ + a] = pm[a];
      out_s [(size_t)row * ACT_ + a] = fminf(fmaxf(ps[a], -20.0f), 2.0f);
    }
    out_v[row] = vacc + bv3[0];
  }
}

extern "C" void kernel_launch(void* const* d_in, const int* in_sizes, int n_in,
                              void* d_out, int out_size, void* d_ws, size_t ws_size,
                              hipStream_t stream)
{
  const float* obs    = (const float*)d_in[0];
  const float* W1     = (const float*)d_in[1];
  const float* b1     = (const float*)d_in[2];
  const float* W2     = (const float*)d_in[3];
  const float* b2     = (const float*)d_in[4];
  const float* Wg1    = (const float*)d_in[5];
  const float* bg1    = (const float*)d_in[6];
  const float* Wg2    = (const float*)d_in[7];
  const float* bg2    = (const float*)d_in[8];
  const float* muW    = (const float*)d_in[9];
  const float* mub    = (const float*)d_in[10];   // [16,6] == flat [96] bias
  const float* lstd   = (const float*)d_in[11];
  const float* Wv1    = (const float*)d_in[12];
  const float* bv1    = (const float*)d_in[13];
  const float* Wv2    = (const float*)d_in[14];
  const float* bv2    = (const float*)d_in[15];
  const float* Wv3    = (const float*)d_in[16];
  const float* bv3    = (const float*)d_in[17];
  const float* cent   = (const float*)d_in[18];
  const float* smean  = (const float*)d_in[19];
  const float* svar   = (const float*)d_in[20];

  const int B = in_sizes[0] / 17;   // 131072

  // ---- workspace carving (256B aligned) ----
  uint8_t* w = (uint8_t*)d_ws;
  size_t o = 0;
  auto take = [&](size_t bytes) -> uint8_t* {
    uint8_t* p = w + o;
    o = (o + bytes + 255) & ~(size_t)255;
    return p;
  };
  _Float16* OBS16 = (_Float16*)take((size_t)B * 32  * 2);
  _Float16* H1    = (_Float16*)take((size_t)B * 256 * 2);   // reused as VH1
  _Float16* FEAT  = (_Float16*)take((size_t)B * 256 * 2);
  _Float16* GT1   = (_Float16*)take((size_t)B * 128 * 2);
  float*    LOGI  = (float*)   take((size_t)B * 16  * 4);
  float*    CD    = (float*)   take((size_t)B * 16  * 4);
  _Float16* MUS   = (_Float16*)take((size_t)B * 96  * 2);
  _Float16* VH2   = (_Float16*)take((size_t)B * 256 * 2);
  _Float16* Wt1   = (_Float16*)take(256 * 32  * 2);
  _Float16* Wt2   = (_Float16*)take(256 * 256 * 2);
  _Float16* Wtg1  = (_Float16*)take(128 * 256 * 2);
  _Float16* Wtg2  = (_Float16*)take(16  * 128 * 2);
  _Float16* WtMU  = (_Float16*)take(96  * 256 * 2);
  _Float16* WtV1  = (_Float16*)take(256 * 256 * 2);
  _Float16* WtV2  = (_Float16*)take(256 * 256 * 2);
  _Float16* WtC   = (_Float16*)take(16  * 256 * 2);
  float*    cc    = (float*)   take(16 * 4);
  float*    offm  = (float*)   take(16 * 4);
  (void)ws_size; (void)n_in; (void)out_size;

  float* out_mu = (float*)d_out;
  float* out_s  = out_mu + (size_t)B * ACT_;
  float* out_v  = out_s  + (size_t)B * ACT_;

  // ---- prep ----
  prep_obs<<<4096, 256, 0, stream>>>(obs, OBS16, B * 32);
  prep_weights<<<256, 256, 0, stream>>>(W1, W2, Wg1, Wg2, muW, Wv1, Wv2, cent,
                                        smean, svar,
                                        Wt1, Wt2, Wtg1, Wtg2, WtMU, WtV1, WtV2, WtC,
                                        cc, offm);

  const int gblocks = B / 128;   // 8 waves x 16 rows per block
  // encoder
  gemm_wmma_kern<1, 16, true,  false><<<gblocks, 256, 0, stream>>>(OBS16, Wt1,  b1,   H1,   256);
  gemm_wmma_kern<8, 16, true,  false><<<gblocks, 256, 0, stream>>>(H1,    Wt2,  b2,   FEAT, 256);
  // gate
  gemm_wmma_kern<8, 8,  true,  false><<<gblocks, 256, 0, stream>>>(FEAT,  Wtg1, bg1,  GT1,  128);
  gemm_wmma_kern<4, 1,  false, true ><<<gblocks, 256, 0, stream>>>(GT1,   Wtg2, bg2,  LOGI, 16);
  // whitened center cross terms: zw . c_m = feat.(c/sig) + offm
  gemm_wmma_kern<8, 1,  false, true ><<<gblocks, 256, 0, stream>>>(FEAT,  WtC,  offm, CD,   16);
  // expert means [B,96]
  gemm_wmma_kern<8, 6,  false, false><<<gblocks, 256, 0, stream>>>(FEAT,  WtMU, mub,  MUS,  96);
  // value MLP (VH1 reuses H1)
  gemm_wmma_kern<8, 16, true,  false><<<gblocks, 256, 0, stream>>>(FEAT,  WtV1, bv1,  H1,   256);
  gemm_wmma_kern<8, 16, true,  false><<<gblocks, 256, 0, stream>>>(H1,    WtV2, bv2,  VH2,  256);

  // per-row epilogue: d2, masked softmax, combine, value dot
  epilogue_kern<<<B / 8, 256, 0, stream>>>(FEAT, LOGI, CD, MUS, VH2, cc,
                                           smean, svar, lstd, Wv3, bv3,
                                           out_mu, out_s, out_v);
}
